// AdaptivePoolingClassifier_44470091383437
// MI455X (gfx1250) — matrix operand order repaired and checked by reading, not compiled, not run on hardware
//
#include <hip/hip_runtime.h>
#include <hip/hip_bf16.h>

typedef __attribute__((ext_vector_type(16))) _Float16 v16h;
typedef __attribute__((ext_vector_type(8)))  float    v8f;

#define D 128
#define ROWS_PER_BLK 128
#define PITCH 136            // halfs per LDS row (128 + 8 pad)
#define NEG_BIG -3.402823466e38f

union FragU { v16h h; uint4 q[2]; };
union H4 { _Float16 h[4]; uint2 u; };

// ---------------- prep kernels (tiny, run once per launch) ----------------

// C[M,N] = A[M,K] @ B[K,N], all f32 row-major
__global__ void gemm_f32_k(const float* __restrict__ A, const float* __restrict__ B,
                           float* __restrict__ C, int M, int K, int N) {
    int idx = blockIdx.x * blockDim.x + threadIdx.x;
    if (idx >= M * N) return;
    int i = idx / N, j = idx % N;
    float s = 0.f;
    for (int k = 0; k < K; ++k) s += A[i * K + k] * B[k * N + j];
    C[idx] = s;
}

// out[j] = b_add[j] + sum_k b_in[k] * B[k,j]
__global__ void biasvec_k(const float* __restrict__ b_in, const float* __restrict__ B,
                          const float* __restrict__ b_add, float* __restrict__ out,
                          int K, int N) {
    int j = blockIdx.x * blockDim.x + threadIdx.x;
    if (j >= N) return;
    float s = b_add[j];
    for (int k = 0; k < K; ++k) s += b_in[k] * B[k * N + j];
    out[j] = s;
}

// Pack Wc[128,128] f32 -> f16 WMMA B-fragments.
// packed[((nt*4+kt)*32+lane)*16 + e] = Wc[(kt*32 + (e<8 ? (lane>>4)*8 + e : 16 + (lane>>4)*8 + e - 8))*128 + nt*16 + (lane&15)]
__global__ void pack_wc_k(const float* __restrict__ Wc, _Float16* __restrict__ packed) {
    int tid = blockIdx.x * blockDim.x + threadIdx.x;   // 16384 threads
    if (tid >= 128 * 128) return;
    int e = tid & 15;
    int lane = (tid >> 4) & 31;
    int kt = (tid >> 9) & 3;
    int nt = tid >> 11;
    int K = kt * 32 + (e < 8 ? (lane >> 4) * 8 + e : 16 + (lane >> 4) * 8 + (e - 8));
    int n = nt * 16 + (lane & 15);
    packed[tid] = (_Float16)Wc[K * 128 + n];
}

// Pack W4[128,5] f32 -> f16 B-fragments for one padded 16-col tile (cols>=5 are 0).
__global__ void pack_w4_k(const float* __restrict__ W4, _Float16* __restrict__ packed) {
    int tid = blockIdx.x * blockDim.x + threadIdx.x;   // 2048 threads
    if (tid >= 4 * 32 * 16) return;
    int e = tid & 15;
    int lane = (tid >> 4) & 31;
    int kt = tid >> 9;
    int K = kt * 32 + (e < 8 ? (lane >> 4) * 8 + e : 16 + (lane >> 4) * 8 + (e - 8));
    int n = lane & 15;
    packed[tid] = (n < 5) ? (_Float16)W4[K * 5 + n] : (_Float16)0.f;
}

// ---------------- main fused MLP kernel ----------------

__global__ __launch_bounds__(256)
void mlp_main_k(const float* __restrict__ x,
                const _Float16* __restrict__ pWc, const float* __restrict__ bc,
                const _Float16* __restrict__ pW4, const float* __restrict__ b4,
                const float* __restrict__ alpha,
                float* __restrict__ p_ws, float* __restrict__ blockmax) {
    __shared__ _Float16 hbuf[ROWS_PER_BLK * PITCH];
    __shared__ float smax[8 * 16];

    const int tid = threadIdx.x;
    const size_t rowbase = (size_t)blockIdx.x * ROWS_PER_BLK;

    // ---- stage x (f32 -> f16) into LDS ----
    const float4* xin = (const float4*)(x + rowbase * D);
    #pragma unroll
    for (int i = 0; i < 16; ++i) {
        int e4 = i * 256 + tid;          // float4 index, 0..4095
        float4 v = xin[e4];
        int el = e4 * 4;
        int row = el >> 7;
        int col = el & 127;
        H4 t;
        t.h[0] = (_Float16)v.x; t.h[1] = (_Float16)v.y;
        t.h[2] = (_Float16)v.z; t.h[3] = (_Float16)v.w;
        *(uint2*)&hbuf[row * PITCH + col] = t.u;
    }
    __syncthreads();

    const int wave = tid >> 5;
    const int lane = tid & 31;
    const int m = lane & 15;             // A row / D col
    const int kb = (lane >> 4) * 8;      // A k-subgroup base
    const int hi = lane >> 4;            // D row group (0 or 1)

    // ---- fused layer Wc: 128 -> 128, wave owns N-tile = wave ----
    v8f zero8 = {0.f, 0.f, 0.f, 0.f, 0.f, 0.f, 0.f, 0.f};
    v8f acc[8];
    #pragma unroll
    for (int mt = 0; mt < 8; ++mt) acc[mt] = zero8;

    const uint4* wq = (const uint4*)pWc;
    #pragma unroll
    for (int kt = 0; kt < 4; ++kt) {
        FragU B;
        int fo = ((wave * 4 + kt) * 32 + lane) * 2;   // uint4 units
        B.q[0] = wq[fo];
        B.q[1] = wq[fo + 1];
        #pragma unroll
        for (int mt = 0; mt < 8; ++mt) {
            FragU A;
            const _Float16* arow = &hbuf[(mt * 16 + m) * PITCH + kt * 32 + kb];
            A.q[0] = *(const uint4*)(arow);
            A.q[1] = *(const uint4*)(arow + 16);
            acc[mt] = __builtin_amdgcn_wmma_f32_16x16x32_f16(
                false, A.h, false, B.h, (short)0, acc[mt], false, false);
        }
    }

    // ---- bias + ReLU, write h back to LDS (in place, after barrier) ----
    const int ncol = wave * 16 + m;
    const float bias = bc[ncol];
    __syncthreads();   // all reads of x staging done
    #pragma unroll
    for (int mt = 0; mt < 8; ++mt) {
        #pragma unroll
        for (int r = 0; r < 8; ++r) {
            float v = acc[mt][r] + bias;
            v = fmaxf(v, 0.f);
            int row = mt * 16 + r + 8 * hi;
            hbuf[row * PITCH + ncol] = (_Float16)v;
        }
    }
    __syncthreads();

    // ---- head W4: 128 -> 5 (padded to 16), wave owns M-tile = wave ----
    v8f acc2 = zero8;
    const uint4* w4q = (const uint4*)pW4;
    const int mt4 = wave;
    #pragma unroll
    for (int kt = 0; kt < 4; ++kt) {
        FragU B;
        int fo = (kt * 32 + lane) * 2;
        B.q[0] = w4q[fo];
        B.q[1] = w4q[fo + 1];
        FragU A;
        const _Float16* arow = &hbuf[(mt4 * 16 + m) * PITCH + kt * 32 + kb];
        A.q[0] = *(const uint4*)(arow);
        A.q[1] = *(const uint4*)(arow + 16);
        acc2 = __builtin_amdgcn_wmma_f32_16x16x32_f16(
            false, A.h, false, B.h, (short)0, acc2, false, false);
    }

    const int col = m;                   // output column 0..15 (valid < 5)
    float a_c = 0.f, b4c = 0.f;
    if (col < 5) { b4c = b4[col]; a_c = alpha[col]; }

    float lmax = NEG_BIG;
    #pragma unroll
    for (int r = 0; r < 8; ++r) {
        float p = acc2[r] + b4c;
        size_t grow = rowbase + (size_t)(mt4 * 16 + r + 8 * hi);
        if (col < 5) p_ws[grow * 5 + col] = p;
        lmax = fmaxf(lmax, a_c * p);
    }
    // combine lane l with l^16 (same column, other 8 rows)
    lmax = fmaxf(lmax, __shfl_xor(lmax, 16, 32));
    if (lane < 16) smax[wave * 16 + lane] = lmax;
    __syncthreads();
    if (tid < 16) {
        float mx = NEG_BIG;
        #pragma unroll
        for (int w = 0; w < 8; ++w) mx = fmaxf(mx, smax[w * 16 + tid]);
        if (tid < 5) blockmax[(size_t)blockIdx.x * 5 + tid] = mx;
    }
}

// ---------------- reduction kernels ----------------

__global__ void gmax_k(const float* __restrict__ blockmax, float* __restrict__ gmax, int nblk) {
    int c = threadIdx.x;
    if (c < 5) {
        float mx = NEG_BIG;
        for (int i = 0; i < nblk; ++i) mx = fmaxf(mx, blockmax[i * 5 + c]);
        gmax[c] = mx;
    }
}

__global__ __launch_bounds__(128)
void partial_k(const float* __restrict__ p_ws, const float* __restrict__ gmax,
               const float* __restrict__ alpha,
               float* __restrict__ bs1, float* __restrict__ bs2) {
    __shared__ float red1[5 * 128];
    __shared__ float red2[5 * 128];
    int tid = threadIdx.x;
    size_t row = (size_t)blockIdx.x * 128 + tid;
    #pragma unroll
    for (int c = 0; c < 5; ++c) {
        float p = p_ws[row * 5 + c];
        float e = __expf(alpha[c] * p - gmax[c]);
        red1[c * 128 + tid] = e;
        red2[c * 128 + tid] = p * e;
    }
    __syncthreads();
    for (int s = 64; s > 0; s >>= 1) {
        if (tid < s) {
            #pragma unroll
            for (int c = 0; c < 5; ++c) {
                red1[c * 128 + tid] += red1[c * 128 + tid + s];
                red2[c * 128 + tid] += red2[c * 128 + tid + s];
            }
        }
        __syncthreads();
    }
    if (tid < 5) {
        bs1[(size_t)blockIdx.x * 5 + tid] = red1[tid * 128];
        bs2[(size_t)blockIdx.x * 5 + tid] = red2[tid * 128];
    }
}

__global__ void finalize_k(const float* __restrict__ bs1, const float* __restrict__ bs2,
                           float* __restrict__ out, int nblk) {
    int c = threadIdx.x;
    if (c < 5) {
        float s1 = 0.f, s2 = 0.f;
        for (int i = 0; i < nblk; ++i) {
            s1 += bs1[i * 5 + c];
            s2 += bs2[i * 5 + c];
        }
        out[c] = s2 / s1;
    }
}

// ---------------- launch ----------------

extern "C" void kernel_launch(void* const* d_in, const int* in_sizes, int n_in,
                              void* d_out, int out_size, void* d_ws, size_t ws_size,
                              hipStream_t stream) {
    const float* x     = (const float*)d_in[0];
    const float* W1    = (const float*)d_in[1];
    const float* b1    = (const float*)d_in[2];
    const float* W2    = (const float*)d_in[3];
    const float* b2    = (const float*)d_in[4];
    const float* W3    = (const float*)d_in[5];
    const float* b3    = (const float*)d_in[6];
    const float* W4    = (const float*)d_in[7];
    const float* b4    = (const float*)d_in[8];
    const float* alpha = (const float*)d_in[9];
    float* out = (float*)d_out;

    const int nrows = in_sizes[0] / D;            // 262144
    const int nblk  = nrows / ROWS_PER_BLK;       // 2048

    // workspace carve-up
    char* wsp = (char*)d_ws;
    size_t off = 0;
    auto carve = [&](size_t bytes) -> void* {
        void* p = wsp + off;
        off = (off + bytes + 255) & ~(size_t)255;
        return p;
    };
    float*     T1   = (float*)    carve(128 * 256 * sizeof(float));
    float*     bt   = (float*)    carve(256 * sizeof(float));
    float*     Wc   = (float*)    carve(128 * 128 * sizeof(float));
    float*     bc   = (float*)    carve(128 * sizeof(float));
    _Float16*  pWc  = (_Float16*) carve(128 * 128 * sizeof(_Float16));
    _Float16*  pW4  = (_Float16*) carve(4 * 32 * 16 * sizeof(_Float16));
    float*     p_ws = (float*)    carve((size_t)nrows * 5 * sizeof(float));
    float*     bmax = (float*)    carve((size_t)nblk * 5 * sizeof(float));
    float*     gmx  = (float*)    carve(8 * sizeof(float));
    float*     bs1  = (float*)    carve((size_t)nblk * 5 * sizeof(float));
    float*     bs2  = (float*)    carve((size_t)nblk * 5 * sizeof(float));

    // 1) fuse linear layers: Wc = W1@W2@W3, bc = b1@W2@W3 + b2@W3 + b3
    gemm_f32_k<<<(128 * 256 + 255) / 256, 256, 0, stream>>>(W1, W2, T1, 128, 384, 256);
    biasvec_k<<<1, 256, 0, stream>>>(b1, W2, b2, bt, 384, 256);
    gemm_f32_k<<<(128 * 128 + 255) / 256, 256, 0, stream>>>(T1, W3, Wc, 128, 256, 128);
    biasvec_k<<<1, 256, 0, stream>>>(bt, W3, b3, bc, 256, 128);

    // 2) pack f16 WMMA B-fragments
    pack_wc_k<<<(128 * 128 + 255) / 256, 256, 0, stream>>>(Wc, pWc);
    pack_w4_k<<<(2048 + 255) / 256, 256, 0, stream>>>(W4, pW4);

    // 3) fused MLP + per-block max
    mlp_main_k<<<nblk, 256, 0, stream>>>(x, pWc, bc, pW4, b4, alpha, p_ws, bmax);

    // 4) softmax-weighted pooling
    gmax_k<<<1, 32, 0, stream>>>(bmax, gmx, nblk);
    partial_k<<<nblk, 128, 0, stream>>>(p_ws, gmx, alpha, bs1, bs2);
    finalize_k<<<1, 32, 0, stream>>>(bs1, bs2, out, nblk);
}
